// HyperTextGuidedVisualBlock_5952824672966
// MI455X (gfx1250) — compile-verified, matrix operand order checked by reading, not run on hardware
//
#include <hip/hip_runtime.h>

// ---------------- types for WMMA ----------------
typedef __attribute__((ext_vector_type(16))) __bf16 v16bf;
typedef __attribute__((ext_vector_type(8)))  float  v8f;
typedef __attribute__((ext_vector_type(8)))  unsigned short v8u;

union Frag16 {
  struct { v8u lo, hi; } u;
  v16bf v;
};

// ---------------- constants (Poincare ball, c = 0.5) ----------------
#define SQC    0.70710678118654752440f           // sqrt(0.5)
#define MAXNRM ((1.0f - 4e-3f) / SQC)            // projection radius
#define CCURV  0.5f

// ---------------- helpers ----------------
__device__ __forceinline__ unsigned short f2bf(float f) {
  union { float f; unsigned u; } v; v.f = f;
  unsigned r = v.u + 0x7FFFu + ((v.u >> 16) & 1u);   // round-to-nearest-even
  return (unsigned short)(r >> 16);
}
__device__ __forceinline__ float bf2f(unsigned short h) {
  union { unsigned u; float f; } v; v.u = ((unsigned)h) << 16;
  return v.f;
}
__device__ __forceinline__ float artanh_c(float z) {
  z = fminf(fmaxf(z, -1.0f + 1e-5f), 1.0f - 1e-5f);
  return 0.5f * logf((1.0f + z) / (1.0f - z));
}
__device__ __forceinline__ float wave_sum(float v) {
  #pragma unroll
  for (int m = 16; m > 0; m >>= 1) v += __shfl_xor(v, m, 32);
  return v;
}
// blockDim.x == 256 (8 wave32's) for every caller
__device__ __forceinline__ float block_sum(float v, float* sm) {
  int tid = threadIdx.x;
  float w = wave_sum(v);
  __syncthreads();                 // protect previous use of sm
  if ((tid & 31) == 0) sm[tid >> 5] = w;
  __syncthreads();
  float tot = 0.f;
  #pragma unroll
  for (int i = 0; i < 8; i++) tot += sm[i];
  return tot;
}

// ---------------- zero fill (u16) ----------------
__global__ __launch_bounds__(256) void k_zero_u16(unsigned short* __restrict__ p, int n) {
  int i = blockIdx.x * 256 + threadIdx.x;
  if (i < n) p[i] = 0;
}

// ---------------- f32 -> bf16 convert ----------------
__global__ __launch_bounds__(256) void k_cvt_bf16(const float* __restrict__ src,
                                                  unsigned short* __restrict__ dst, int n) {
  int i = blockIdx.x * 256 + threadIdx.x;
  if (i < n) dst[i] = f2bf(src[i]);
}

// ---------------- expmap0 + project, row length N==blockDim (text path) ----------------
__global__ __launch_bounds__(256) void k_expmap_cvt(const float* __restrict__ X,
                                                    float* __restrict__ outF,
                                                    unsigned short* __restrict__ outB,
                                                    float* __restrict__ outN, int N) {
  __shared__ float sm[8];
  int row = blockIdx.x, t = threadIdx.x;
  float u = X[(size_t)row * N + t];
  float ss = block_sum(u * u, sm);
  float n = fmaxf(sqrtf(ss), 1e-15f);
  float s = fminf(tanhf(SQC * n) / (SQC * n), MAXNRM / n);  // expmap scale + project
  float y = u * s;
  outF[(size_t)row * N + t] = y;
  outB[(size_t)row * N + t] = f2bf(y);
  if (t == 0) outN[row] = n * s;
}

// ---------------- fused expmap0 (over W rows) + patchify -> bf16 ----------------
// one wave per (b,c,d,h) row of 128 floats; patch row n=(gd,gh,gw), col=c*256+pd*64+ph*8+pw
__global__ __launch_bounds__(256) void k_expmap_patchify(const float* __restrict__ vf,
                                                         unsigned short* __restrict__ Xp) {
  int gwave = blockIdx.x * 8 + (threadIdx.x >> 5);
  int lane  = threadIdx.x & 31;
  int h = gwave & 127;
  int d = (gwave >> 7) & 63;
  int c = (gwave >> 13) & 7;
  int b = gwave >> 16;
  const float* row = vf + (size_t)gwave * 128;
  float4 u = *(const float4*)(row + lane * 4);
  float ss = wave_sum(u.x * u.x + u.y * u.y + u.z * u.z + u.w * u.w);
  float n = fmaxf(sqrtf(ss), 1e-15f);
  float s = fminf(tanhf(SQC * n) / (SQC * n), MAXNRM / n);
  size_t base = ((size_t)b * 4096 + (size_t)(d >> 2) * 256 + (size_t)(h >> 3) * 16) * 2048
              + (size_t)c * 256 + (size_t)(d & 3) * 64 + (size_t)(h & 7) * 8;
  float vals[4] = {u.x, u.y, u.z, u.w};
  #pragma unroll
  for (int i = 0; i < 4; i++) {
    int w = lane * 4 + i;
    Xp[base + (size_t)(w >> 3) * 2048 + (w & 7)] = f2bf(vals[i] * s);
  }
}

// ---------------- row L2 norm over bf16 matrix ----------------
__global__ __launch_bounds__(256) void k_rownorm_bf16(const unsigned short* __restrict__ X,
                                                      float* __restrict__ norms, int K) {
  __shared__ float sm[8];
  int row = blockIdx.x;
  const unsigned short* p = X + (size_t)row * K;
  float ss = 0.f;
  for (int i = threadIdx.x; i < K; i += 256) { float v = bf2f(p[i]); ss += v * v; }
  ss = block_sum(ss, sm);
  if (threadIdx.x == 0) norms[row] = sqrtf(ss);
}

// ---------------- row norm + bf16 convert of an f32 matrix (N==256) ----------------
__global__ __launch_bounds__(256) void k_rownorm_cvt(const float* __restrict__ X,
                                                     float* __restrict__ norms,
                                                     unsigned short* __restrict__ outB, int N) {
  __shared__ float sm[8];
  int row = blockIdx.x, t = threadIdx.x;
  float v = X[(size_t)row * N + t];
  float ss = block_sum(v * v, sm);
  outB[(size_t)row * N + t] = f2bf(v);
  if (t == 0) norms[row] = sqrtf(ss);
}

// ---------------- bf16 WMMA GEMM: C[M,N] = A[M,K] * B[N,K]^T ----------------
// Requirements: M % 128 == 0, N % 64 == 0, K % 32 == 0 (caller pads).
// block = 256 thr (8 waves); tile 128x64; each wave: 2x2 WMMA tiles of 16x16.
// Double-buffered LDS: one barrier per k-step; next tile's global loads issued
// before the current tile's WMMAs.
__global__ __launch_bounds__(256) void k_gemm_bf16(const unsigned short* __restrict__ A,
                                                   const unsigned short* __restrict__ B,
                                                   float* __restrict__ C,
                                                   int M, int N, int K) {
  __shared__ __align__(16) unsigned short As[2][128 * 32];
  __shared__ __align__(16) unsigned short Bs[2][64 * 32];
  int tid  = threadIdx.x;
  int wave = tid >> 5, lane = tid & 31;
  int bm = blockIdx.y * 128, bn = blockIdx.x * 64;
  int mi0 = (wave & 3) * 2;     // m-tile pair: 8 m-tiles across 4 wave groups
  int nj0 = (wave >> 2) * 2;    // n-tile pair: 4 n-tiles across 2 wave groups
  int half = lane >> 4;         // ISA 16-bit A/B layout: lanes 16-31 hold K+8
  int l16  = lane & 15;
  int klo  = half ? 8 : 0;
  int khi  = half ? 24 : 16;
  // staging coords: A tile 128x32 = 512 v8u (2/thread), B tile 64x32 = 256 v8u (1/thread)
  int ar = tid >> 1;             // 0..127
  int ac = (tid & 1) * 16;       // 0 or 16 (two adjacent v8u)
  int br = tid >> 2;             // 0..63
  int bc = (tid & 3) * 8;        // 0,8,16,24
  const unsigned short* Ap = A + (size_t)bm * K;
  const unsigned short* Bp = B + (size_t)bn * K;

  v8f acc00 = {}, acc01 = {}, acc10 = {}, acc11 = {};

  // preload k-tile 0
  v8u ra0 = *(const v8u*)(Ap + (size_t)ar * K + ac);
  v8u ra1 = *(const v8u*)(Ap + (size_t)ar * K + ac + 8);
  v8u rb  = *(const v8u*)(Bp + (size_t)br * K + bc);
  *(v8u*)(As[0] + ar * 32 + ac)     = ra0;
  *(v8u*)(As[0] + ar * 32 + ac + 8) = ra1;
  *(v8u*)(Bs[0] + br * 32 + bc)     = rb;
  __syncthreads();

  int nk = K >> 5;
  for (int kt = 0; kt < nk; kt++) {
    int cur = kt & 1, nxt = cur ^ 1;
    if (kt + 1 < nk) {
      int ko = (kt + 1) << 5;
      ra0 = *(const v8u*)(Ap + (size_t)ar * K + ko + ac);
      ra1 = *(const v8u*)(Ap + (size_t)ar * K + ko + ac + 8);
      rb  = *(const v8u*)(Bp + (size_t)br * K + ko + bc);
      *(v8u*)(As[nxt] + ar * 32 + ac)     = ra0;
      *(v8u*)(As[nxt] + ar * 32 + ac + 8) = ra1;
      *(v8u*)(Bs[nxt] + br * 32 + bc)     = rb;
    }
    const unsigned short* as = As[cur];
    const unsigned short* bs = Bs[cur];
    Frag16 fa0, fa1, fb0, fb1;
    fa0.u.lo = *(const v8u*)(as + (mi0 * 16 + l16) * 32 + klo);
    fa0.u.hi = *(const v8u*)(as + (mi0 * 16 + l16) * 32 + khi);
    fa1.u.lo = *(const v8u*)(as + ((mi0 + 1) * 16 + l16) * 32 + klo);
    fa1.u.hi = *(const v8u*)(as + ((mi0 + 1) * 16 + l16) * 32 + khi);
    fb0.u.lo = *(const v8u*)(bs + (nj0 * 16 + l16) * 32 + klo);
    fb0.u.hi = *(const v8u*)(bs + (nj0 * 16 + l16) * 32 + khi);
    fb1.u.lo = *(const v8u*)(bs + ((nj0 + 1) * 16 + l16) * 32 + klo);
    fb1.u.hi = *(const v8u*)(bs + ((nj0 + 1) * 16 + l16) * 32 + khi);
    acc00 = __builtin_amdgcn_wmma_f32_16x16x32_bf16(false, fa0.v, false, fb0.v,
                                                    (short)0, acc00, false, false);
    acc01 = __builtin_amdgcn_wmma_f32_16x16x32_bf16(false, fa0.v, false, fb1.v,
                                                    (short)0, acc01, false, false);
    acc10 = __builtin_amdgcn_wmma_f32_16x16x32_bf16(false, fa1.v, false, fb0.v,
                                                    (short)0, acc10, false, false);
    acc11 = __builtin_amdgcn_wmma_f32_16x16x32_bf16(false, fa1.v, false, fb1.v,
                                                    (short)0, acc11, false, false);
    __syncthreads();
  }
  // C/D layout: lane<16 -> rows r, lane>=16 -> rows r+8; col = lane&15
  int col0 = bn + nj0 * 16 + l16;
  int col1 = col0 + 16;
  int row0 = bm + mi0 * 16 + half * 8;
  int row1 = row0 + 16;
  #pragma unroll
  for (int r = 0; r < 8; r++) {
    C[(size_t)(row0 + r) * N + col0] = acc00[r];
    C[(size_t)(row0 + r) * N + col1] = acc01[r];
    C[(size_t)(row1 + r) * N + col0] = acc10[r];
    C[(size_t)(row1 + r) * N + col1] = acc11[r];
  }
}

// ---------------- mobius_matvec rescale + mobius_add(bias) + project (N==256) ----------------
__global__ __launch_bounds__(256) void k_hyp_post(const float* __restrict__ mx,
                                                  const float* __restrict__ xn,
                                                  const float* __restrict__ bias,
                                                  float* __restrict__ outF,
                                                  unsigned short* __restrict__ outB,
                                                  float* __restrict__ outN, int N) {
  __shared__ float sm[8];
  int row = blockIdx.x, t = threadIdx.x;
  float m  = mx[(size_t)row * N + t];
  float bv = bias[t];
  float ss  = block_sum(m * m, sm);
  float mxn = fmaxf(sqrtf(ss), 1e-15f);
  float x   = fmaxf(xn[row], 1e-15f);
  float sc  = tanhf((mxn / x) * artanh_c(SQC * x)) / (mxn * SQC);
  float res = (ss > 0.f) ? sc * m : 0.f;      // all-zero mx -> zero vector
  float nr  = sc * mxn;
  if (ss > 0.f && nr > MAXNRM) res *= MAXNRM / nr;   // project
  // mobius_add(res, bias)
  float x2 = block_sum(res * res, sm);
  float y2 = block_sum(bv * bv, sm);
  float xy = block_sum(res * bv, sm);
  float num = (1.f + 2.f * CCURV * xy + CCURV * y2) * res + (1.f - CCURV * x2) * bv;
  float den = fmaxf(1.f + 2.f * CCURV * xy + CCURV * CCURV * x2 * y2, 1e-15f);
  float o = num / den;
  float on = fmaxf(sqrtf(block_sum(o * o, sm)), 1e-15f);
  if (on > MAXNRM) o *= MAXNRM / on;                  // project
  if (outF) outF[(size_t)row * N + t] = o;
  if (outB) outB[(size_t)row * N + t] = f2bf(o);
  if (outN && t == 0) outN[row] = fminf(on, MAXNRM);
}

// ---------------- Poincare-distance attention: one wave per (b,q,head) ----------------
__global__ __launch_bounds__(256) void k_attention(const float* __restrict__ Q,  // (B*Nq,256)
                                                   const float* __restrict__ Kf, // (32,256)
                                                   const float* __restrict__ Vf, // (32,256)
                                                   float* __restrict__ O) {      // (B*Nq,256)
  const int E = 256, HD = 64, NK = 32;
  int gw   = blockIdx.x * 8 + (threadIdx.x >> 5);
  int lane = threadIdx.x & 31;
  int h = gw & 3;
  int q = (gw >> 2) & 4095;
  int b = gw >> 14;
  const float* qp = Q + ((size_t)b * 4096 + q) * E + h * HD;
  float q0 = qp[lane * 2], q1 = qp[lane * 2 + 1];
  float q2 = fminf(wave_sum(q0 * q0 + q1 * q1), 1.f - 1e-5f);
  float my_score = 0.f;                         // score for key == lane
  for (int kk = 0; kk < NK; kk++) {
    const float* kp = Kf + (size_t)kk * E + h * HD;
    float k0 = kp[lane * 2], k1 = kp[lane * 2 + 1];
    float kq = wave_sum(q0 * k0 + q1 * k1);
    float k2 = fminf(wave_sum(k0 * k0 + k1 * k1), 1.f - 1e-5f);
    float den   = (1.f - q2) * (1.f - k2) + 1e-5f;
    float delta = 2.f * (kq - q2 * k2) / den;
    float dist  = sqrtf(fmaxf(delta, 1e-5f));
    if (lane == kk) my_score = -dist;
  }
  float mxs = my_score;
  #pragma unroll
  for (int m = 16; m > 0; m >>= 1) mxs = fmaxf(mxs, __shfl_xor(mxs, m, 32));
  float e = expf(my_score - mxs);
  float p = e / wave_sum(e);
  float o0 = 0.f, o1 = 0.f;
  for (int kk = 0; kk < NK; kk++) {
    float pk = __shfl(p, kk, 32);
    const float* vp = Vf + (size_t)kk * E + h * HD;
    o0 += pk * vp[lane * 2];
    o1 += pk * vp[lane * 2 + 1];
  }
  float* op = O + ((size_t)b * 4096 + q) * E + h * HD;
  op[lane * 2] = o0; op[lane * 2 + 1] = o1;
}

// ---------------- logmap0 -> LayerNorm -> expmap0 -> logmap0 (N==256) ----------------
__global__ __launch_bounds__(256) void k_ln_tangent(const float* __restrict__ X,
                                                    const float* __restrict__ gamma,
                                                    const float* __restrict__ beta,
                                                    float* __restrict__ outT) {
  const int N = 256;
  __shared__ float sm[8];
  int row = blockIdx.x, t = threadIdx.x;
  float y = X[(size_t)row * N + t];
  float n = fmaxf(sqrtf(block_sum(y * y, sm)), 1e-15f);
  float tl = artanh_c(SQC * n) / (SQC * n) * y;            // logmap0
  float mu = block_sum(tl, sm) * (1.f / N);
  float d  = tl - mu;
  float var = block_sum(d * d, sm) * (1.f / N);
  float tn = d * rsqrtf(var + 1e-5f) * gamma[t] + beta[t];
  float n2 = fmaxf(sqrtf(block_sum(tn * tn, sm)), 1e-15f); // expmap0 + project
  float s2 = fminf(tanhf(SQC * n2) / (SQC * n2), MAXNRM / n2);
  float ny = fmaxf(n2 * s2, 1e-15f);
  float s3 = artanh_c(SQC * ny) / (SQC * ny);              // logmap0 of result
  outT[(size_t)row * N + t] = s3 * s2 * tn;
}

// ---------------- 1x1x1 conv E->8: one wave per (b, token) ----------------
__global__ __launch_bounds__(256) void k_conv(const float* __restrict__ T,   // (B*4096,256)
                                              const float* __restrict__ Wc,  // (8,256)
                                              const float* __restrict__ bc,  // (8)
                                              float* __restrict__ F) {       // (B,8,4096)
  int gw   = blockIdx.x * 8 + (threadIdx.x >> 5);
  int lane = threadIdx.x & 31;
  int n = gw & 4095, b = gw >> 12;
  const float* tp = T + ((size_t)b * 4096 + n) * 256;
  float acc[8] = {0, 0, 0, 0, 0, 0, 0, 0};
  for (int e = lane; e < 256; e += 32) {
    float tv = tp[e];
    #pragma unroll
    for (int o = 0; o < 8; o++) acc[o] += Wc[o * 256 + e] * tv;
  }
  float mine = 0.f;
  #pragma unroll
  for (int o = 0; o < 8; o++) {
    float s = wave_sum(acc[o]);
    if (lane == o) mine = s;
  }
  if (lane < 8) F[((size_t)b * 8 + lane) * 4096 + n] = mine + bc[lane];
}

// ---------------- trilinear resize (16,16,16)->(64,128,128) + blend ----------------
__global__ __launch_bounds__(256) void k_resize_blend(const float* __restrict__ vf,
                                                      const float* __restrict__ FS,
                                                      const float* __restrict__ alpha,
                                                      float* __restrict__ out) {
  size_t i = (size_t)blockIdx.x * 256 + threadIdx.x;
  int w = (int)(i & 127);
  int h = (int)((i >> 7) & 127);
  int d = (int)((i >> 14) & 63);
  int c = (int)((i >> 20) & 7);
  int b = (int)(i >> 23);
  float fz = (d + 0.5f) * 0.25f  - 0.5f;
  float fy = (h + 0.5f) * 0.125f - 0.5f;
  float fx = (w + 0.5f) * 0.125f - 0.5f;
  int z0 = (int)floorf(fz), y0 = (int)floorf(fy), x0 = (int)floorf(fx);
  float wz = fz - z0, wy = fy - y0, wx = fx - x0;
  int z1 = z0 + 1 > 15 ? 15 : z0 + 1;
  int y1 = y0 + 1 > 15 ? 15 : y0 + 1;
  int x1 = x0 + 1 > 15 ? 15 : x0 + 1;
  z0 = z0 < 0 ? 0 : z0; y0 = y0 < 0 ? 0 : y0; x0 = x0 < 0 ? 0 : x0;
  const float* base = FS + ((size_t)b * 8 + c) * 4096;
  #define G(z, y, x) base[(z) * 256 + (y) * 16 + (x)]
  float c00 = G(z0, y0, x0) * (1.f - wx) + G(z0, y0, x1) * wx;
  float c01 = G(z0, y1, x0) * (1.f - wx) + G(z0, y1, x1) * wx;
  float c10 = G(z1, y0, x0) * (1.f - wx) + G(z1, y0, x1) * wx;
  float c11 = G(z1, y1, x0) * (1.f - wx) + G(z1, y1, x1) * wx;
  #undef G
  float f = (c00 * (1.f - wy) + c01 * wy) * (1.f - wz)
          + (c10 * (1.f - wy) + c11 * wy) * wz;
  float a = 1.f / (1.f + expf(-alpha[0]));
  float v = vf[i];
  out[i] = v + a * (f - v);
}

// ================= host-side launch =================
extern "C" void kernel_launch(void* const* d_in, const int* in_sizes, int n_in,
                              void* d_out, int out_size, void* d_ws, size_t ws_size,
                              hipStream_t stream) {
  (void)in_sizes; (void)n_in; (void)out_size; (void)ws_size;
  const float* vf    = (const float*)d_in[0];
  const float* text  = (const float*)d_in[1];
  const float* Wp    = (const float*)d_in[2];
  const float* bp    = (const float*)d_in[3];
  const float* Wq    = (const float*)d_in[4];
  const float* bq    = (const float*)d_in[5];
  const float* Wk    = (const float*)d_in[6];
  const float* bk    = (const float*)d_in[7];
  const float* Wv    = (const float*)d_in[8];
  const float* bv    = (const float*)d_in[9];
  const float* Wo    = (const float*)d_in[10];
  const float* bo    = (const float*)d_in[11];
  const float* gamma = (const float*)d_in[12];
  const float* beta  = (const float*)d_in[13];
  const float* convw = (const float*)d_in[14];
  const float* convb = (const float*)d_in[15];
  const float* alpha = (const float*)d_in[16];
  float* out = (float*)d_out;

  const size_t M = 16384, E = 256, Kp = 2048;
  const size_t MT = 128;                 // padded text rows (one M-tile)
  char* w = (char*)d_ws;
  size_t off = 0;
  auto alloc = [&](size_t bytes) { size_t o = off; off += (bytes + 255) & ~(size_t)255; return o; };
  const size_t o_big  = alloc(M * Kp * 2);        // 64MB: Xp bf16, later 4x16MB f32 aliases
  const size_t o_xn   = alloc(M * 4);
  const size_t o_wpb  = alloc(E * Kp * 2);
  const size_t o_wqb  = alloc(E * E * 2);
  const size_t o_wkb  = alloc(E * E * 2);
  const size_t o_wvb  = alloc(E * E * 2);
  const size_t o_wob  = alloc(E * E * 2);
  const size_t o_txtf = alloc(32 * E * 4);
  const size_t o_txtb = alloc(MT * E * 2);        // zero-padded to 128 rows
  const size_t o_txtn = alloc(32 * 4);
  const size_t o_mx   = alloc(M * E * 4);         // GEMM output scratch (16MB, reused)
  const size_t o_tokb = alloc(M * E * 2);         // tok bf16, reused as attn-out bf16
  const size_t o_tokn = alloc(M * 4);             // tok norms, reused as attn-out norms
  const size_t o_kf   = alloc(32 * E * 4);
  const size_t o_vf   = alloc(32 * E * 4);
  const size_t o_fs   = alloc(4 * 8 * 4096 * 4);
  // aliases inside o_big (Xp dead after the Wp GEMM)
  const size_t o_qf  = o_big;
  const size_t o_aof = o_big + 1 * (M * E * 4);
  const size_t o_o2f = o_big + 2 * (M * E * 4);
  const size_t o_tt  = o_big + 3 * (M * E * 4);

  unsigned short* XP   = (unsigned short*)(w + o_big);
  float*          XN   = (float*)(w + o_xn);
  unsigned short* WPB  = (unsigned short*)(w + o_wpb);
  unsigned short* WQB  = (unsigned short*)(w + o_wqb);
  unsigned short* WKB  = (unsigned short*)(w + o_wkb);
  unsigned short* WVB  = (unsigned short*)(w + o_wvb);
  unsigned short* WOB  = (unsigned short*)(w + o_wob);
  float*          TXTF = (float*)(w + o_txtf);
  unsigned short* TXTB = (unsigned short*)(w + o_txtb);
  float*          TXTN = (float*)(w + o_txtn);
  float*          MX   = (float*)(w + o_mx);
  unsigned short* TOKB = (unsigned short*)(w + o_tokb);
  float*          TOKN = (float*)(w + o_tokn);
  float*          KF   = (float*)(w + o_kf);
  float*          VF   = (float*)(w + o_vf);
  float*          FS   = (float*)(w + o_fs);
  float*          QF   = (float*)(w + o_qf);
  float*          AOF  = (float*)(w + o_aof);
  float*          O2F  = (float*)(w + o_o2f);
  float*          TT   = (float*)(w + o_tt);

  // 1) weights -> bf16
  k_cvt_bf16<<<dim3((unsigned)((E * Kp) / 256)), 256, 0, stream>>>(Wp, WPB, (int)(E * Kp));
  k_cvt_bf16<<<dim3((unsigned)((E * E) / 256)), 256, 0, stream>>>(Wq, WQB, (int)(E * E));
  k_cvt_bf16<<<dim3((unsigned)((E * E) / 256)), 256, 0, stream>>>(Wk, WKB, (int)(E * E));
  k_cvt_bf16<<<dim3((unsigned)((E * E) / 256)), 256, 0, stream>>>(Wv, WVB, (int)(E * E));
  k_cvt_bf16<<<dim3((unsigned)((E * E) / 256)), 256, 0, stream>>>(Wo, WOB, (int)(E * E));
  // 2) text expmap0 (zero-pad to 128 rows first so the GEMM needs no guards)
  k_zero_u16<<<dim3((unsigned)((MT * E) / 256)), 256, 0, stream>>>(TXTB, (int)(MT * E));
  k_expmap_cvt<<<32, 256, 0, stream>>>(text, TXTF, TXTB, TXTN, 256);
  // 3) visual expmap0 + patchify (streams 128MB once)
  k_expmap_patchify<<<32768, 256, 0, stream>>>(vf, XP);
  // 4) patch row norms
  k_rownorm_bf16<<<16384, 256, 0, stream>>>(XP, XN, 2048);
  // 5) tok = hyp_linear(Xp, Wp, bp)
  k_gemm_bf16<<<dim3(4, 128), 256, 0, stream>>>(XP, WPB, MX, 16384, 256, 2048);
  k_hyp_post<<<16384, 256, 0, stream>>>(MX, XN, bp, nullptr, TOKB, TOKN, 256);
  // 6) q = hyp_linear(tok, Wq, bq)
  k_gemm_bf16<<<dim3(4, 128), 256, 0, stream>>>(TOKB, WQB, MX, 16384, 256, 256);
  k_hyp_post<<<16384, 256, 0, stream>>>(MX, TOKN, bq, QF, nullptr, nullptr, 256);
  // 7) k,v from text (batch-invariant; padded rows compute zeros, discarded)
  k_gemm_bf16<<<dim3(4, 1), 256, 0, stream>>>(TXTB, WKB, MX, (int)MT, 256, 256);
  k_hyp_post<<<32, 256, 0, stream>>>(MX, TXTN, bk, KF, nullptr, nullptr, 256);
  k_gemm_bf16<<<dim3(4, 1), 256, 0, stream>>>(TXTB, WVB, MX, (int)MT, 256, 256);
  k_hyp_post<<<32, 256, 0, stream>>>(MX, TXTN, bv, VF, nullptr, nullptr, 256);
  // 8) hyperbolic attention
  k_attention<<<8192, 256, 0, stream>>>(QF, KF, VF, AOF);
  // 9) attn-out norms + bf16 (reuse TOKN/TOKB)
  k_rownorm_cvt<<<16384, 256, 0, stream>>>(AOF, TOKN, TOKB, 256);
  // 10) out = hyp_linear(attn_out, Wo, bo)
  k_gemm_bf16<<<dim3(4, 128), 256, 0, stream>>>(TOKB, WOB, MX, 16384, 256, 256);
  k_hyp_post<<<16384, 256, 0, stream>>>(MX, TOKN, bo, O2F, nullptr, nullptr, 256);
  // 11) logmap0 -> LN -> expmap0 -> logmap0
  k_ln_tangent<<<16384, 256, 0, stream>>>(O2F, gamma, beta, TT);
  // 12) 1x1x1 conv E->8
  k_conv<<<2048, 256, 0, stream>>>(TT, convw, convb, FS);
  // 13) trilinear upsample + sigmoid(alpha) blend (streams 256MB)
  k_resize_blend<<<131072, 256, 0, stream>>>(vf, FS, alpha, out);
}
